// SSM_cha_56427280335364
// MI455X (gfx1250) — matrix operand order repaired and checked by reading, not compiled, not run on hardware
//
#include <hip/hip_runtime.h>
#include <hip/hip_bf16.h>

typedef __attribute__((ext_vector_type(16))) __bf16 v16bf;
typedef __attribute__((ext_vector_type(8)))  float  v8f;

#define B_N    4
#define L_N    4096
#define DM     64
#define DIN    128
#define DSTATE 16
#define NROWS  (B_N * L_N)     // 16384
#define LDXD   48              // padded xdbl width (36 valid)

// ---------------------------------------------------------------------------
// C[M, Npad] = A[M,K] (fp32, row-major) @ W[nvalid, K]^T  via v_wmma bf16.
// One wave -> one 16x16 tile; 8 waves/block. K is compile-time (full unroll),
// GUARD=true handles Npad > nvalid (x-proj padding) branchlessly.
// ---------------------------------------------------------------------------
template <int K, bool GUARD>
__global__ void gemm_wmma_bf16(const float* __restrict__ A,
                               const float* __restrict__ W,
                               float* __restrict__ C,
                               int M, int Npad, int nvalid) {
    const int lane = threadIdx.x & 31;
    const int wave = threadIdx.x >> 5;
    const int gw   = blockIdx.x * 8 + wave;
    const int tilesN = Npad >> 4;
    const int tm = gw / tilesN;
    const int tn = gw - tm * tilesN;
    const int mBase = tm << 4;
    const int nBase = tn << 4;
    if (mBase >= M) return;

    const int l15   = lane & 15;
    const int kgrpA = (lane & 16) ? 8 : 0;
    const int kgrpB = (lane & 16) ? 8 : 0;
    const float* ap = A + (size_t)(mBase + l15) * K + kgrpA;

    const int  wr  = nBase + l15;
    const bool wok = !GUARD || (wr < nvalid);
    const float* wp = W + (size_t)(wok ? wr : 0) * K + kgrpB;
    const float wsc = wok ? 1.f : 0.f;

    v8f acc = {};
#pragma unroll
    for (int k0 = 0; k0 < K; k0 += 32) {
        const float4 a0 = *reinterpret_cast<const float4*>(ap + k0);
        const float4 a1 = *reinterpret_cast<const float4*>(ap + k0 + 4);
        const float4 a2 = *reinterpret_cast<const float4*>(ap + k0 + 16);
        const float4 a3 = *reinterpret_cast<const float4*>(ap + k0 + 20);
        float4 b0 = *reinterpret_cast<const float4*>(wp + k0);
        float4 b1 = *reinterpret_cast<const float4*>(wp + k0 + 4);
        float4 b2 = *reinterpret_cast<const float4*>(wp + k0 + 16);
        float4 b3 = *reinterpret_cast<const float4*>(wp + k0 + 20);
        if (GUARD) {
            b0.x *= wsc; b0.y *= wsc; b0.z *= wsc; b0.w *= wsc;
            b1.x *= wsc; b1.y *= wsc; b1.z *= wsc; b1.w *= wsc;
            b2.x *= wsc; b2.y *= wsc; b2.z *= wsc; b2.w *= wsc;
            b3.x *= wsc; b3.y *= wsc; b3.z *= wsc; b3.w *= wsc;
        }
        v16bf a, b;
        a[0]  = (__bf16)a0.x; a[1]  = (__bf16)a0.y; a[2]  = (__bf16)a0.z; a[3]  = (__bf16)a0.w;
        a[4]  = (__bf16)a1.x; a[5]  = (__bf16)a1.y; a[6]  = (__bf16)a1.z; a[7]  = (__bf16)a1.w;
        a[8]  = (__bf16)a2.x; a[9]  = (__bf16)a2.y; a[10] = (__bf16)a2.z; a[11] = (__bf16)a2.w;
        a[12] = (__bf16)a3.x; a[13] = (__bf16)a3.y; a[14] = (__bf16)a3.z; a[15] = (__bf16)a3.w;
        b[0]  = (__bf16)b0.x; b[1]  = (__bf16)b0.y; b[2]  = (__bf16)b0.z; b[3]  = (__bf16)b0.w;
        b[4]  = (__bf16)b1.x; b[5]  = (__bf16)b1.y; b[6]  = (__bf16)b1.z; b[7]  = (__bf16)b1.w;
        b[8]  = (__bf16)b2.x; b[9]  = (__bf16)b2.y; b[10] = (__bf16)b2.z; b[11] = (__bf16)b2.w;
        b[12] = (__bf16)b3.x; b[13] = (__bf16)b3.y; b[14] = (__bf16)b3.z; b[15] = (__bf16)b3.w;
        acc = __builtin_amdgcn_wmma_f32_16x16x32_bf16(
            /*neg_a=*/false, a, /*neg_b=*/false, b,
            /*c_mod=*/(short)0, acc, /*reuse_a=*/false, /*reuse_b=*/false);
    }

    const int col   = nBase + l15;
    const int rbase = mBase + ((lane & 16) ? 8 : 0);
    if (!GUARD || col < nvalid) {
#pragma unroll
        for (int i = 0; i < 8; ++i)
            C[(size_t)(rbase + i) * Npad + col] = acc[i];
    }
}

// ---------------------------------------------------------------------------
// (B,C,H,W) -> (B, L, C)
// ---------------------------------------------------------------------------
__global__ void transpose_in(const float* __restrict__ x, float* __restrict__ o) {
    int idx = blockIdx.x * blockDim.x + threadIdx.x;
    if (idx >= B_N * L_N * DM) return;
    int c = idx & (DM - 1);
    int r = idx >> 6;              // b*L + l
    int b = r >> 12;
    int l = r & (L_N - 1);
    o[idx] = x[((size_t)(b * DM + c)) * L_N + l];
}

// (B, L, C) -> (B,C,H,W)
__global__ void transpose_out(const float* __restrict__ x, float* __restrict__ o) {
    int idx = blockIdx.x * blockDim.x + threadIdx.x;
    if (idx >= B_N * DM * L_N) return;
    int l = idx & (L_N - 1);
    int r = idx >> 12;             // b*DM + c
    int b = r >> 6;
    int c = r & (DM - 1);
    o[idx] = x[((size_t)(b * L_N + l)) * DM + c];
}

// ---------------------------------------------------------------------------
// Depthwise causal conv (k=4) over L within each batch + SiLU (branchless).
// XR: (NROWS, 256), u-part at cols [0,128). U out: (NROWS, 128).
// ---------------------------------------------------------------------------
__global__ void conv_silu(const float* __restrict__ XR,
                          const float* __restrict__ cw,   // (128,1,4)
                          const float* __restrict__ cb,   // (128)
                          float* __restrict__ U) {
    int idx = blockIdx.x * blockDim.x + threadIdx.x;
    if (idx >= B_N * L_N * DIN) return;
    int d = idx & (DIN - 1);
    int r = idx >> 7;              // b*L + l
    int b = r >> 12;
    int l = r & (L_N - 1);
    float s = cb[d];
#pragma unroll
    for (int k = 0; k < 4; ++k) {
        const int li = l - 3 + k;
        const int lc = li < 0 ? 0 : li;                   // clamp: load always valid
        const float xv = XR[((size_t)(b * L_N + lc)) * 256 + d];
        s += (li >= 0 ? cw[d * 4 + k] : 0.f) * xv;        // branchless select
    }
    U[idx] = s / (1.f + __expf(-s));                      // silu
}

// ---------------------------------------------------------------------------
// Fused selective scan: one block per batch, one thread per channel d.
// 16 states in registers; dt/B/C staged via double-buffered LDS (one barrier
// per step). Software-pipelined: step t+1's globals are loaded while step t's
// 16 v_exp chain executes; global_prefetch pulls rows 16 ahead into WGP$.
// Applies +u*D and silu(res) gating inline.
// ---------------------------------------------------------------------------
__global__ void ssm_scan(const float* __restrict__ XD,   // (NROWS,48): [dt4|B16|C16]
                         const float* __restrict__ U,    // (NROWS,128)
                         const float* __restrict__ XR,   // (NROWS,256): res at 128+d
                         const float* __restrict__ dtw,  // (128,4)
                         const float* __restrict__ dtb,  // (128)
                         const float* __restrict__ Alog, // (128,16)
                         const float* __restrict__ Dp,   // (128)
                         float* __restrict__ Y) {        // (NROWS,128)
    __shared__ float lds[2 * LDXD];
    const int b = blockIdx.x;
    const int d = threadIdx.x;
    const int row0 = b * L_N;

    float Aa[DSTATE], h[DSTATE], w[4];
#pragma unroll
    for (int n = 0; n < DSTATE; ++n) {
        Aa[n] = -__expf(Alog[d * DSTATE + n]);
        h[n]  = 0.f;
    }
#pragma unroll
    for (int r = 0; r < 4; ++r) w[r] = dtw[d * 4 + r];
    const float bdt = dtb[d];
    const float Dd  = Dp[d];

    // prologue: fetch t=0 operands
    float xd_n  = (d < 36) ? XD[(size_t)row0 * LDXD + d] : 0.f;
    float u_n   = U[(size_t)row0 * DIN + d];
    float res_n = XR[(size_t)row0 * 256 + DIN + d];

    for (int t = 0; t < L_N; ++t) {
        const int row  = row0 + t;
        const int slot = (t & 1) * LDXD;
        if (d < 36) lds[slot + d] = xd_n;
        __syncthreads();

        const float u   = u_n;
        const float res = res_n;

        // issue next step's loads now; they drain behind the exp chain below
        const int rowp = row + ((t + 1 < L_N) ? 1 : 0);
        if (d < 36) xd_n = XD[(size_t)rowp * LDXD + d];
        u_n   = U[(size_t)rowp * DIN + d];
        res_n = XR[(size_t)rowp * 256 + DIN + d];
        if (t + 16 < L_N) {                       // stream rows into WGP$
            __builtin_prefetch(&U[(size_t)(row + 16) * DIN + d], 0, 0);
            __builtin_prefetch(&XR[(size_t)(row + 16) * 256 + DIN + d], 0, 0);
            if (d < 36) __builtin_prefetch(&XD[(size_t)(row + 16) * LDXD + d], 0, 0);
        }

        float v = bdt;
#pragma unroll
        for (int r = 0; r < 4; ++r) v += lds[slot + r] * w[r];
        const float dt = (v > 20.f) ? v : __logf(1.f + __expf(v));   // softplus

        float y = 0.f;
#pragma unroll
        for (int n = 0; n < DSTATE; ++n) {
            const float dA = __expf(dt * Aa[n]);
            h[n] = dA * h[n] + dt * lds[slot + 4 + n] * u;
            y   += h[n] * lds[slot + 20 + n];
        }
        const float g = res / (1.f + __expf(-res));                  // silu(res)
        Y[(size_t)row * DIN + d] = (y + u * Dd) * g;
    }
}

// ---------------------------------------------------------------------------
// LayerNorm over last dim (64). One wave32 per row, 2 cols/lane, shfl reduce.
// ---------------------------------------------------------------------------
__global__ void layernorm64(const float* __restrict__ X,
                            const float* __restrict__ g,
                            const float* __restrict__ bb,
                            float* __restrict__ O) {
    const int lane = threadIdx.x & 31;
    const int wave = threadIdx.x >> 5;
    const int row  = blockIdx.x * 8 + wave;
    if (row >= NROWS) return;
    const float* xp = X + (size_t)row * 64;
    float x0 = xp[lane], x1 = xp[lane + 32];
    float s  = x0 + x1;
    float s2 = x0 * x0 + x1 * x1;
#pragma unroll
    for (int off = 16; off; off >>= 1) {
        s  += __shfl_xor(s,  off, 32);
        s2 += __shfl_xor(s2, off, 32);
    }
    const float m = s * (1.f / 64.f);
    const float v = s2 * (1.f / 64.f) - m * m;
    const float r = rsqrtf(v + 1e-5f);
    O[(size_t)row * 64 + lane]      = (x0 - m) * r * g[lane]      + bb[lane];
    O[(size_t)row * 64 + lane + 32] = (x1 - m) * r * g[lane + 32] + bb[lane + 32];
}

// ---------------------------------------------------------------------------
extern "C" void kernel_launch(void* const* d_in, const int* in_sizes, int n_in,
                              void* d_out, int out_size, void* d_ws, size_t ws_size,
                              hipStream_t stream) {
    (void)in_sizes; (void)n_in; (void)out_size; (void)ws_size;
    const float* x1 = (const float*)d_in[0];

    struct BlkW {
        const float *in_w, *conv_w, *conv_b, *xproj_w, *dt_w, *dt_b, *Alog, *D, *out_w, *ln_g, *ln_b;
    };
    BlkW bw[2];
    bw[0] = { (const float*)d_in[1], (const float*)d_in[2], (const float*)d_in[3],
              (const float*)d_in[4], (const float*)d_in[5], (const float*)d_in[6],
              (const float*)d_in[7], (const float*)d_in[8], (const float*)d_in[9],
              (const float*)d_in[19], (const float*)d_in[20] };
    bw[1] = { (const float*)d_in[10], (const float*)d_in[11], (const float*)d_in[12],
              (const float*)d_in[13], (const float*)d_in[14], (const float*)d_in[15],
              (const float*)d_in[16], (const float*)d_in[17], (const float*)d_in[18],
              (const float*)d_in[21], (const float*)d_in[22] };

    float* ws = (float*)d_ws;
    float* X0 = ws;                         // 16384*64   (activation / LN out)
    float* XR = X0 + (size_t)NROWS * 64;    // 16384*256  (in-proj out: u|res)
    float* U  = XR + (size_t)NROWS * 256;   // 16384*128  (conv+silu out)
    float* XD = U  + (size_t)NROWS * 128;   // 16384*48   (x-proj out, padded)
    float* Y  = XD + (size_t)NROWS * LDXD;  // 16384*128  (scan+gate out)
    float* T  = Y  + (size_t)NROWS * 128;   // 16384*64   (out-proj out)

    transpose_in<<<(NROWS * DM) / 256, 256, 0, stream>>>(x1, X0);

    for (int i = 0; i < 2; ++i) {
        // in-proj: (16384,64) x (256,64)^T -> (16384,256)   K=64
        gemm_wmma_bf16<64, false><<<(NROWS / 16) * (256 / 16) / 8, 256, 0, stream>>>(
            X0, bw[i].in_w, XR, NROWS, 256, 256);
        // depthwise conv + silu -> U
        conv_silu<<<(NROWS * DIN) / 256, 256, 0, stream>>>(XR, bw[i].conv_w, bw[i].conv_b, U);
        // x-proj: (16384,128) x (36,128)^T -> (16384,48 padded, 36 valid)
        gemm_wmma_bf16<128, true><<<(NROWS / 16) * (LDXD / 16) / 8, 256, 0, stream>>>(
            U, bw[i].xproj_w, XD, NROWS, LDXD, 36);
        // fused selective scan + D skip + silu(res) gating
        ssm_scan<<<B_N, DIN, 0, stream>>>(XD, U, XR, bw[i].dt_w, bw[i].dt_b,
                                          bw[i].Alog, bw[i].D, Y);
        // out-proj: (16384,128) x (64,128)^T -> (16384,64)  K=128
        gemm_wmma_bf16<128, false><<<(NROWS / 16) * (64 / 16) / 8, 256, 0, stream>>>(
            Y, bw[i].out_w, T, NROWS, 64, 64);
        // layernorm -> X0 (feeds next block / final transpose)
        layernorm64<<<NROWS / 8, 256, 0, stream>>>(T, bw[i].ln_g, bw[i].ln_b, X0);
    }

    transpose_out<<<(NROWS * DM) / 256, 256, 0, stream>>>(X0, (float*)d_out);
}
